// ITNSR_3478923510157
// MI455X (gfx1250) — matrix operand order repaired and checked by reading, not compiled, not run on hardware
//
#include <hip/hip_runtime.h>

// ---------------------------------------------------------------------------
// ITNSR for MI455X (gfx1250).  95% of FLOPs = imnet MLP GEMMs (M=131072,
// K=256, N=256/576) -> bf16 WMMA, activations LDS-resident, weights streamed
// via TDM (tensor_load_to_lds) with double buffering.  RDN backbone (5% of
// FLOPs) kept fp32 VALU.  128 rows/block to halve L2 weight re-streaming.
// ---------------------------------------------------------------------------

#define HH 48
#define WW 48
#define HW 2304
#define QN 32768
#define NROW 131072            // 4 corners * QN

typedef __attribute__((ext_vector_type(16))) __bf16  v16bf;
typedef __attribute__((ext_vector_type(8)))  float   v8f;
typedef __attribute__((ext_vector_type(4)))  unsigned int u32x4;
typedef __attribute__((ext_vector_type(4)))  int     i32x4;
typedef __attribute__((ext_vector_type(8)))  int     i32x8;

// -------------------------- small helpers ----------------------------------
__device__ __forceinline__ float gelu_f(float x) {
    return 0.5f * x * (1.0f + erff(x * 0.7071067811865476f));   // exact erf GELU
}
__device__ __forceinline__ unsigned short f2bf(float f) {       // RNE f32->bf16
    unsigned u = __float_as_uint(f);
    u += 0x7FFFu + ((u >> 16) & 1u);
    return (unsigned short)(u >> 16);
}
__device__ __forceinline__ float bf2f(unsigned short h) {
    return __uint_as_float(((unsigned)h) << 16);
}

// ----------------------- RDN: generic conv (fp32) --------------------------
__global__ void k_conv(const float* __restrict__ x, const float* __restrict__ w,
                       const float* __restrict__ b, float* __restrict__ y,
                       int Cin, int Cout, int ksize, int relu,
                       const float* __restrict__ add)
{
    int t = blockIdx.x * blockDim.x + threadIdx.x;
    if (t >= Cout * HW) return;
    int o = t / HW, p = t % HW;
    int py = p / WW, px = p % WW;
    float acc = b[o];
    if (ksize == 1) {
        const float* wr = w + o * Cin;
        for (int i = 0; i < Cin; ++i) acc += wr[i] * x[i * HW + p];
    } else {
        const float* wr = w + o * Cin * 9;
        for (int i = 0; i < Cin; ++i)
            for (int ki = 0; ki < 3; ++ki) {
                int yy = py + ki - 1;
                if ((unsigned)yy >= (unsigned)HH) continue;
                for (int kj = 0; kj < 3; ++kj) {
                    int xx = px + kj - 1;
                    if ((unsigned)xx >= (unsigned)WW) continue;
                    acc += wr[(i * 3 + ki) * 3 + kj] * x[i * HW + yy * WW + xx];
                }
            }
    }
    if (add) acc += add[t];
    if (relu) acc = fmaxf(acc, 0.0f);
    y[t] = acc;
}

// -------------------- featU: unfold(3x3) -> bf16, [pixel][576] -------------
__global__ void k_featU(const float* __restrict__ feat, unsigned short* __restrict__ fU)
{
    int t = blockIdx.x * blockDim.x + threadIdx.x;
    if (t >= HW * 576) return;
    int p = t / 576, ch = t % 576;
    int c = ch / 9, k = ch % 9, ki = k / 3, kj = k % 3;
    int py = p / WW + ki - 1, px = p % WW + kj - 1;
    float v = 0.0f;
    if ((unsigned)py < (unsigned)HH && (unsigned)px < (unsigned)WW)
        v = feat[c * HW + py * WW + px];
    fU[p * 576 + ch] = f2bf(v);
}

// ---------------- transpose+convert imnet weights: Wt[n][k] bf16 -----------
__global__ void k_wt(const float* __restrict__ W, unsigned short* __restrict__ Wt,
                     int K, int N)
{
    int t = blockIdx.x * blockDim.x + threadIdx.x;
    if (t >= K * N) return;
    int n = t % N, k = t / N;
    Wt[n * K + k] = f2bf(W[k * N + n]);
}

// ----------------------- per-query/corner prep -----------------------------
__global__ void k_prep(const float* __restrict__ coord, const float* __restrict__ scale,
                       float* __restrict__ X0, int* __restrict__ IDX,
                       float* __restrict__ REL)
{
    int t = blockIdx.x * blockDim.x + threadIdx.x;
    if (t >= NROW) return;
    int corner = t >> 15, q = t & (QN - 1);
    float vx = (corner & 2) ? 1.0f : -1.0f;
    float vy = (corner & 1) ? 1.0f : -1.0f;
    float rx = (1.0f - scale[0]) / (float)(HH - 1);
    float ry = (1.0f - scale[1]) / (float)(WW - 1);
    float c0 = coord[q * 2 + 0], c1 = coord[q * 2 + 1];
    float cy = fminf(fmaxf(c0 + vx * rx + 1e-6f, -1.0f + 1e-6f), 1.0f - 1e-6f);
    float cx = fminf(fmaxf(c1 + vy * ry + 1e-6f, -1.0f + 1e-6f), 1.0f - 1e-6f);
    int iy = (int)fminf(fmaxf(rintf(((cy + 1.0f) * (float)HH - 1.0f) * 0.5f), 0.0f), (float)(HH - 1));
    int ix = (int)fminf(fmaxf(rintf(((cx + 1.0f) * (float)WW - 1.0f) * 0.5f), 0.0f), (float)(WW - 1));
    float fy = -1.0f + (2.0f * iy + 1.0f) / (float)HH;
    float fx = -1.0f + (2.0f * ix + 1.0f) / (float)WW;
    float r0 = (c0 - fy) * (float)HH;
    float r1 = (c1 - fx) * (float)WW;
    IDX[t] = iy * WW + ix;
    REL[t * 2 + 0] = r0;  REL[t * 2 + 1] = r1;
    X0[t * 4 + 0] = r0;   X0[t * 4 + 1] = r1;
    X0[t * 4 + 2] = scale[q * 2 + 0] * (float)HH;
    X0[t * 4 + 3] = scale[q * 2 + 1] * (float)WW;
}

// ------------------------------ TDM helpers --------------------------------
#if __has_builtin(__builtin_amdgcn_tensor_load_to_lds)
#define HAVE_TDM 1
#else
#define HAVE_TDM 0
#endif

__device__ __forceinline__ void tdm_load_wchunk(const unsigned short* gsrc,
                                                unsigned lds_byte_off,
                                                unsigned tensor_rows)
{
#if HAVE_TDM
    unsigned long long ga = (unsigned long long)(const void*)gsrc;
    u32x4 g0;
    g0[0] = 1u;                                                  // count=1, user D#
    g0[1] = lds_byte_off;                                        // LDS dest (bytes)
    g0[2] = (unsigned)ga;                                        // global addr lo
    g0[3] = (unsigned)((ga >> 32) & 0x01FFFFFFull) | (2u << 30); // addr hi | type=2
    i32x8 g1;
    // data_size=2B, pad_enable, pad every 128 dwords (512B row), pad 4 dwords
    g1[0] = (int)((1u << 16) | (1u << 20) | (6u << 22) | (3u << 25));
    g1[1] = (int)(256u << 16);                         // tensor_dim0 = 256 (K)
    g1[2] = (int)((tensor_rows & 0xFFFFu) << 16);      // tensor_dim1 lo
    g1[3] = (int)((tensor_rows >> 16) | (256u << 16)); // tensor_dim1 hi | tile_dim0=256
    g1[4] = 64;                                        // tile_dim1 = 64 rows
    g1[5] = 256;                                       // tensor_dim0_stride = 256
    g1[6] = 0;
    g1[7] = 0;
    i32x4 z4 = {0, 0, 0, 0};
#if defined(__clang_major__) && (__clang_major__ >= 23)
    i32x8 z8 = {0, 0, 0, 0, 0, 0, 0, 0};
    __builtin_amdgcn_tensor_load_to_lds(g0, g1, z4, z4, z8, 0);
#else
    __builtin_amdgcn_tensor_load_to_lds(g0, g1, z4, z4, 0);
#endif
#endif
}

__device__ __forceinline__ void wait_tensor_le1()
{
#if __has_builtin(__builtin_amdgcn_s_wait_tensorcnt)
    __builtin_amdgcn_s_wait_tensorcnt(1);
#else
    asm volatile("s_wait_tensorcnt 0x1" ::: "memory");
#endif
}
__device__ __forceinline__ void wait_tensor_le0()
{
#if __has_builtin(__builtin_amdgcn_s_wait_tensorcnt)
    __builtin_amdgcn_s_wait_tensorcnt(0);
#else
    asm volatile("s_wait_tensorcnt 0x0" ::: "memory");
#endif
}

// --------------------------- fused MLP kernel ------------------------------
// 128 rows/block, 8 waves (wave w owns M-tile w).  LDS arena (ushort units):
#define ROWS 128
#define ASTR 272                        // act stride (256 + 16 pad)
#define WSTR 264                        // weight stride (256 + 8 pad = 528B, 16B aligned)
#define OFF_ACTA 0
#define OFF_ACTB (ROWS * ASTR)          // 34816 us
#define OFF_WLS0 (2 * ROWS * ASTR)      // 69632 us
#define OFF_WLS1 (OFF_WLS0 + 64 * WSTR) // 86528 us
#define OFF_WLS0_BYTES (OFF_WLS0 * 2)   // 139264
#define OFF_WLS1_BYTES (OFF_WLS1 * 2)   // 173056
#define OFF_IDX_BYTES  (OFF_WLS1_BYTES + 64 * WSTR * 2)   // 206848
#define SMEM_BYTES (OFF_IDX_BYTES + ROWS * 4 + ROWS * 4)  // 207872
#define NCHUNK 21                       // 4+4+4 (layers 1-3) + 9 (layer 4)

// A fragment: 16x32 bf16; lane L holds row (L&15), K halves at base (L&16?8:0)
__device__ __forceinline__ v16bf load_afrag(const unsigned short* act, int mrow,
                                            int k0, int lane)
{
    const unsigned short* p = act + (mrow + (lane & 15)) * ASTR + k0 + ((lane & 16) ? 8 : 0);
    union { v16bf v; uint4 u[2]; } f;
    f.u[0] = *(const uint4*)(p);
    f.u[1] = *(const uint4*)(p + 16);
    return f.v;
}
// B fragment: 32x16 bf16 from Wt[n][k]; lane L holds col (L&15), K halves at base (L&16?8:0)
__device__ __forceinline__ v16bf load_bfrag(const unsigned short* wls, int ntl,
                                            int k0, int lane)
{
    const unsigned short* p = wls + (ntl * 16 + (lane & 15)) * WSTR + k0 + ((lane & 16) ? 8 : 0);
    union { v16bf v; uint4 u[2]; } f;
    f.u[0] = *(const uint4*)(p);
    f.u[1] = *(const uint4*)(p + 16);
    return f.v;
}

// flat weight-chunk schedule across all 4 WMMA layers
__device__ __forceinline__ const unsigned short* chunk_src(
    int t, const unsigned short* W1, const unsigned short* W2,
    const unsigned short* W3, const unsigned short* W4, unsigned& rows)
{
    if (t < 12) {
        rows = 256u;
        const unsigned short* base = (t < 4) ? W1 : (t < 8) ? W2 : W3;
        return base + (size_t)(t & 3) * (64 * 256);
    }
    rows = 576u;
    return W4 + (size_t)(t - 12) * (64 * 256);
}

__global__ void __launch_bounds__(256)
k_mlp(const float* __restrict__ X0, const int* __restrict__ IDX,
      const float* __restrict__ REL, const unsigned short* __restrict__ fU,
      const float* __restrict__ W0, const float* __restrict__ b0,
      const unsigned short* __restrict__ Wt1, const unsigned short* __restrict__ Wt2,
      const unsigned short* __restrict__ Wt3, const unsigned short* __restrict__ Wt4,
      const float* __restrict__ b1, const float* __restrict__ b2,
      const float* __restrict__ b3, const float* __restrict__ b4,
      const float* __restrict__ Ws0, const float* __restrict__ bs0,
      const float* __restrict__ Ws1, const float* __restrict__ bs1,
      float* __restrict__ preds)
{
    extern __shared__ unsigned short smem[];
    unsigned short* actA = smem + OFF_ACTA;
    unsigned short* actB = smem + OFF_ACTB;
    unsigned short* wls0 = smem + OFF_WLS0;
    unsigned short* wls1 = smem + OFF_WLS1;
    int*   sIdx  = (int*)((char*)smem + OFF_IDX_BYTES);
    float* sPred = (float*)((char*)smem + OFF_IDX_BYTES + ROWS * 4);

    const int tid  = threadIdx.x;
    const int wave = tid >> 5, lane = tid & 31;
    const int row0 = blockIdx.x * ROWS;
    const int mrow = wave * 16;                 // this wave's M-tile (fixed)

    if (tid < ROWS) sIdx[tid] = IDX[row0 + tid];

#if HAVE_TDM
    if (wave == 0) {                            // prefetch weight chunk 0
        unsigned rows;
        const unsigned short* src = chunk_src(0, Wt1, Wt2, Wt3, Wt4, rows);
        tdm_load_wchunk(src, OFF_WLS0_BYTES, rows);
    }
#endif

    // ---- layer 0 (4 -> 256), VALU (K=4 too small for WMMA) ----
    {
        int r = tid >> 1, j0 = (tid & 1) * 128;
        float x0 = X0[(row0 + r) * 4 + 0], x1 = X0[(row0 + r) * 4 + 1];
        float x2 = X0[(row0 + r) * 4 + 2], x3 = X0[(row0 + r) * 4 + 3];
        for (int j = j0; j < j0 + 128; ++j) {
            float h = b0[j] + x0 * W0[j] + x1 * W0[256 + j] + x2 * W0[512 + j] + x3 * W0[768 + j];
            actA[r * ASTR + j] = f2bf(gelu_f(h));
        }
    }
    __syncthreads();

    unsigned short* cur = actA;
    unsigned short* nxt = actB;
    float pacc[8];
#pragma unroll
    for (int i = 0; i < 8; ++i) pacc[i] = 0.0f;

    int t = 0;                                   // flat weight-chunk index
    for (int layer = 1; layer <= 4; ++layer) {
        const float* gB = (layer == 1) ? b1 : (layer == 2) ? b2
                          : (layer == 3) ? b3 : b4;
        const int Nl = (layer == 4) ? 576 : 256;

        for (int ch = 0; ch < Nl / 64; ++ch, ++t) {
#if HAVE_TDM
            if (wave == 0) {                     // double-buffered TDM stream
                if (t + 1 < NCHUNK) {
                    unsigned rows;
                    const unsigned short* src = chunk_src(t + 1, Wt1, Wt2, Wt3, Wt4, rows);
                    tdm_load_wchunk(src, ((t + 1) & 1) ? OFF_WLS1_BYTES : OFF_WLS0_BYTES, rows);
                    wait_tensor_le1();           // chunk t landed (in-order TDM)
                } else {
                    wait_tensor_le0();
                }
            }
#else
            {
                unsigned rows;
                const unsigned short* src = chunk_src(t, Wt1, Wt2, Wt3, Wt4, rows);
                unsigned short* wb = (t & 1) ? wls1 : wls0;
                for (int e = tid; e < 64 * 256; e += 256)
                    wb[(e >> 8) * WSTR + (e & 255)] = src[e];
            }
#endif
            __syncthreads();                     // chunk t visible to all waves
            const unsigned short* wb = (t & 1) ? wls1 : wls0;

            v8f c[4];
#pragma unroll
            for (int j = 0; j < 4; ++j) {
                float bv = gB[ch * 64 + j * 16 + (lane & 15)];
#pragma unroll
                for (int i = 0; i < 8; ++i) c[j][i] = bv;
            }
#pragma unroll
            for (int k0 = 0; k0 < 256; k0 += 32) {
                v16bf a = load_afrag(cur, mrow, k0, lane);   // 1 A feeds 4 WMMAs
#pragma unroll
                for (int j = 0; j < 4; ++j) {
                    v16bf bfr = load_bfrag(wb, j, k0, lane);
                    c[j] = __builtin_amdgcn_wmma_f32_16x16x32_bf16(false, a, false, bfr,
                                                                   (short)0, c[j], false, false);
                }
            }
            const int rb = mrow + ((lane & 16) ? 8 : 0);
            if (layer < 4) {                     // GELU -> next activation buffer
#pragma unroll
                for (int j = 0; j < 4; ++j) {
                    int nc = ch * 64 + j * 16 + (lane & 15);
#pragma unroll
                    for (int i = 0; i < 8; ++i)
                        nxt[(rb + i) * ASTR + nc] = f2bf(gelu_f(c[j][i]));
                }
            } else {                             // fused value . weight dot
#pragma unroll
                for (int j = 0; j < 4; ++j) {
                    int nc = ch * 64 + j * 16 + (lane & 15);
#pragma unroll
                    for (int i = 0; i < 8; ++i)
                        pacc[i] += bf2f(fU[(size_t)sIdx[rb + i] * 576 + nc]) * c[j][i];
                }
            }
            __syncthreads();                     // done reading wb before reuse at t+2
        }
        if (layer < 4) { unsigned short* tmp = cur; cur = nxt; nxt = tmp; }
    }

    // cross-lane reduce: 16 lanes share each row set; waves own disjoint rows
#pragma unroll
    for (int i = 0; i < 8; ++i) {
        float v = pacc[i];
        v += __shfl_xor(v, 1, 32);
        v += __shfl_xor(v, 2, 32);
        v += __shfl_xor(v, 4, 32);
        v += __shfl_xor(v, 8, 32);
        if ((lane & 15) == 0)
            sPred[mrow + ((lane & 16) ? 8 : 0) + i] = v;
    }
    __syncthreads();

    // epilogue: + score MLP (2->256->1, exact GELU), write preds[q*4+corner]
    if (tid < ROWS) {
        int r = row0 + tid;
        float r0 = REL[r * 2 + 0], r1 = REL[r * 2 + 1];
        float s = bs1[0];
        for (int j = 0; j < 256; ++j) {
            float h = gelu_f(r0 * Ws0[j] + r1 * Ws0[256 + j] + bs0[j]);
            s += h * Ws1[j];
        }
        int q = r & (QN - 1), corner = r >> 15;
        preds[q * 4 + corner] = sPred[tid] + s;
    }
}

// ------------------------ final 4->256->1 MLP ------------------------------
__global__ void k_final(const float* __restrict__ preds,
                        const float* __restrict__ Ww0, const float* __restrict__ bw0,
                        const float* __restrict__ Ww1, const float* __restrict__ bw1,
                        float* __restrict__ out)
{
    int q = blockIdx.x * blockDim.x + threadIdx.x;
    if (q >= QN) return;
    float p0 = preds[q * 4 + 0], p1 = preds[q * 4 + 1];
    float p2 = preds[q * 4 + 2], p3 = preds[q * 4 + 3];
    float s = bw1[0];
    for (int j = 0; j < 256; ++j) {
        float h = gelu_f(p0 * Ww0[j] + p1 * Ww0[256 + j] +
                         p2 * Ww0[512 + j] + p3 * Ww0[768 + j] + bw0[j]);
        s += h * Ww1[j];
    }
    out[q] = s;
}

// ------------------------------- host side ---------------------------------
static void conv(const float* x, const float* w, const float* b, float* y,
                 int Cin, int Cout, int k, int relu, const float* add, hipStream_t s)
{
    int n = Cout * HW;
    k_conv<<<(n + 255) / 256, 256, 0, s>>>(x, w, b, y, Cin, Cout, k, relu, add);
}

extern "C" void kernel_launch(void* const* d_in, const int* in_sizes, int n_in,
                              void* d_out, int out_size, void* d_ws, size_t ws_size,
                              hipStream_t stream)
{
    (void)in_sizes; (void)n_in; (void)out_size; (void)ws_size;
    const float* inp   = (const float*)d_in[0];
    const float* coord = (const float*)d_in[1];
    const float* scale = (const float*)d_in[2];
    auto P = [&](int i) { return (const float*)d_in[i]; };

    char* ws = (char*)d_ws;
    float* F1   = (float*)(ws + 0);                 //  64*HW f32
    float* FX   = (float*)(ws + 589824);            //  64*HW f32
    float* YB   = (float*)(ws + 1179648);           // 128*HW f32
    float* OUTS = (float*)(ws + 2359296);           // 1024*HW f32
    float* TMP  = (float*)(ws + 11796480);          //  64*HW f32
    unsigned short* FU = (unsigned short*)(ws + 12386304);   // HW*576 bf16
    float* X0b  = (float*)(ws + 15040512);          // NROW*4 f32
    int*   IDXb = (int*)(ws + 17137664);            // NROW i32
    float* RELb = (float*)(ws + 17661952);          // NROW*2 f32
    float* PRb  = (float*)(ws + 18710528);          // NROW f32
    unsigned short* WT1 = (unsigned short*)(ws + 19234816);  // 256*256 bf16 each
    unsigned short* WT2 = WT1 + 65536;
    unsigned short* WT3 = WT2 + 65536;
    unsigned short* WT4 = WT3 + 65536;              // 576*256 bf16

    // ---- RDN backbone ----
    conv(inp, P(3), P(4), F1, 1, 64, 3, 0, nullptr, stream);     // sfe1
    conv(F1, P(5), P(6), FX, 64, 64, 3, 0, nullptr, stream);     // sfe2
    for (int d = 0; d < 16; ++d) {
        int base = 7 + d * 18;
        hipMemcpyAsync(YB, FX, 64 * HW * sizeof(float), hipMemcpyDeviceToDevice, stream);
        for (int c = 0; c < 8; ++c)
            conv(YB, P(base + 2 * c), P(base + 2 * c + 1), YB + (64 + 8 * c) * HW,
                 64 + 8 * c, 8, 3, 1, nullptr, stream);
        float* slot = OUTS + d * 64 * HW;
        conv(YB, P(base + 16), P(base + 17), slot, 128, 64, 1, 0, FX, stream);  // lff + x
        hipMemcpyAsync(FX, slot, 64 * HW * sizeof(float), hipMemcpyDeviceToDevice, stream);
    }
    conv(OUTS, P(295), P(296), TMP, 1024, 64, 1, 0, nullptr, stream);  // gff1
    conv(TMP, P(297), P(298), FX, 64, 64, 3, 0, F1, stream);           // gff2 + f1

    // ---- featU (bf16, [pixel][576]) ----
    k_featU<<<(HW * 576 + 255) / 256, 256, 0, stream>>>(FX, FU);

    // ---- imnet weights -> bf16 transposed [N][K] ----
    k_wt<<<(256 * 256 + 255) / 256, 256, 0, stream>>>(P(301), WT1, 256, 256);
    k_wt<<<(256 * 256 + 255) / 256, 256, 0, stream>>>(P(303), WT2, 256, 256);
    k_wt<<<(256 * 256 + 255) / 256, 256, 0, stream>>>(P(305), WT3, 256, 256);
    k_wt<<<(256 * 576 + 255) / 256, 256, 0, stream>>>(P(307), WT4, 256, 576);

    // ---- query/corner prep ----
    k_prep<<<(NROW + 255) / 256, 256, 0, stream>>>(coord, scale, X0b, IDXb, RELb);

    // ---- fused WMMA MLP + value dot + score (double-buffered TDM stream) ----
    hipFuncSetAttribute((const void*)k_mlp,
                        hipFuncAttributeMaxDynamicSharedMemorySize, SMEM_BYTES);
    k_mlp<<<NROW / ROWS, 256, SMEM_BYTES, stream>>>(
        X0b, IDXb, RELb, FU,
        P(299), P(300), WT1, WT2, WT3, WT4,
        P(302), P(304), P(306), P(308),
        P(313), P(314), P(315), P(316), PRb);

    // ---- final weight MLP ----
    k_final<<<(QN + 255) / 256, 256, 0, stream>>>(PRb, P(309), P(310), P(311), P(312),
                                                  (float*)d_out);
}
